// Attention_6158983102614
// MI455X (gfx1250) — matrix operand order repaired and checked by reading, not compiled, not run on hardware
//
#include <hip/hip_runtime.h>
#include <hip/hip_bf16.h>
#include <math.h>

typedef __attribute__((ext_vector_type(16))) __bf16 bf16x16;
typedef __attribute__((ext_vector_type(8)))  float  floatx8;

__device__ __forceinline__ floatx8 zero8() {
    floatx8 z = {0.f, 0.f, 0.f, 0.f, 0.f, 0.f, 0.f, 0.f};
    return z;
}

__device__ __forceinline__ unsigned short f2bf(float f) {
    __bf16 h = (__bf16)f;
    return __builtin_bit_cast(unsigned short, h);
}

// ---------------------------------------------------------------- conversions
__global__ void cvt_f32_bf16(const float* __restrict__ in,
                             unsigned short* __restrict__ out, int n) {
    for (int i = blockIdx.x * blockDim.x + threadIdx.x; i < n;
         i += gridDim.x * blockDim.x)
        out[i] = f2bf(in[i]);
}

// in[K][N] (fp32, row-major) -> out[N][K] (bf16)
__global__ void transpose_cvt(const float* __restrict__ in,
                              unsigned short* __restrict__ out, int K, int N) {
    int total = K * N;
    for (int i = blockIdx.x * blockDim.x + threadIdx.x; i < total;
         i += gridDim.x * blockDim.x) {
        int kidx = i / N, nidx = i % N;
        out[(size_t)nidx * K + kidx] = f2bf(in[i]);
    }
}

// ---------------------------------------------------------------- QKV GEMM
// xb [M=8192][K=768] bf16 ; wT [2304][768] bf16 (w_qkv transposed)
// 32x64 output tile per wave (2 A-frags share 4 B-frags -> 8 WMMA / k-chunk).
// Scatter to q/k [B,H,N,64] and vT [B,H,64,N].
__global__ __launch_bounds__(32)
void qkv_gemm(const unsigned short* __restrict__ xb,
              const unsigned short* __restrict__ wT,
              unsigned short* __restrict__ q,
              unsigned short* __restrict__ kmat,
              unsigned short* __restrict__ vT) {
    const int lane = threadIdx.x;
    const int lp = lane & 15, hf = lane >> 4;
    const int mtile = blockIdx.x;   // 256
    const int ntile = blockIdx.y;   // 36
    const int K = 768;

    floatx8 acc[2][4];
#pragma unroll
    for (int g = 0; g < 2; ++g)
#pragma unroll
        for (int t = 0; t < 4; ++t) acc[g][t] = zero8();

    const unsigned short* arow0 = xb + (size_t)(mtile * 32 + lp) * K;
    const unsigned short* arow1 = arow0 + (size_t)16 * K;
    const unsigned short* brow[4];
#pragma unroll
    for (int t = 0; t < 4; ++t)
        brow[t] = wT + (size_t)(ntile * 64 + t * 16 + lp) * K;

    for (int kk = 0; kk < K; kk += 32) {
        if (kk + 32 < K) {
            __builtin_prefetch(arow0 + kk + 32, 0, 0);
            __builtin_prefetch(arow1 + kk + 32, 0, 0);
        }
        bf16x16 a0, a1;  // A 16x32: per-lane runs at half*8 and 16+half*8
        ((uint4*)&a0)[0] = *(const uint4*)(arow0 + kk + hf * 8);
        ((uint4*)&a0)[1] = *(const uint4*)(arow0 + kk + 16 + hf * 8);
        ((uint4*)&a1)[0] = *(const uint4*)(arow1 + kk + hf * 8);
        ((uint4*)&a1)[1] = *(const uint4*)(arow1 + kk + 16 + hf * 8);
#pragma unroll
        for (int t = 0; t < 4; ++t) {
            bf16x16 bm;  // B 32x16: per-lane runs at half*8 and 16+half*8
            ((uint4*)&bm)[0] = *(const uint4*)(brow[t] + kk + hf * 8);
            ((uint4*)&bm)[1] = *(const uint4*)(brow[t] + kk + 16 + hf * 8);
            acc[0][t] = __builtin_amdgcn_wmma_f32_16x16x32_bf16(
                false, a0, false, bm, (short)0, acc[0][t], false, false);
            acc[1][t] = __builtin_amdgcn_wmma_f32_16x16x32_bf16(
                false, a1, false, bm, (short)0, acc[1][t], false, false);
        }
    }

    const int which = ntile / 12, h = ntile % 12;  // 64-col group == one head
#pragma unroll
    for (int g = 0; g < 2; ++g)
#pragma unroll
        for (int t = 0; t < 4; ++t)
#pragma unroll
            for (int j = 0; j < 8; ++j) {
                int m = mtile * 32 + g * 16 + j + 8 * hf;  // row in [0,8192)
                int bidx = m >> 11, nseq = m & 2047;
                int d = t * 16 + lp;
                unsigned short bits = f2bf(acc[g][t][j]);
                size_t bh = (size_t)(bidx * 12 + h);
                if (which == 0)
                    q[(bh * 2048 + nseq) * 64 + d] = bits;
                else if (which == 1)
                    kmat[(bh * 2048 + nseq) * 64 + d] = bits;
                else
                    vT[(bh * 64 + d) * 2048 + nseq] = bits;
            }
}

// ---------------------------------------------------------------- flash attn
// One wave per (b, h, 32-row query tile): two 16-row groups share every
// K-fragment and V-fragment (halves K/V L2 traffic). Online softmax; P goes
// C-layout -> A-layout through a 2KB LDS staging buffer.
__global__ __launch_bounds__(32)
void flash_attn(const unsigned short* __restrict__ q,
                const unsigned short* __restrict__ kmat,
                const unsigned short* __restrict__ vT,
                unsigned short* __restrict__ y) {
    const int lane = threadIdx.x;
    const int lp = lane & 15, hf = lane >> 4;
    const int mtile = blockIdx.x;          // 64
    const int h = blockIdx.y, b = blockIdx.z;
    const size_t bh = (size_t)b * 12 + h;
    const float scale = 0.125f;            // 64^-0.5

    __shared__ __align__(16) unsigned short pbuf[2 * 16 * 32];

    // Q tile 32x64 as 2x2 A-fragments, resident for the whole loop
    bf16x16 aq[2][2];
#pragma unroll
    for (int g = 0; g < 2; ++g) {
        const unsigned short* qrow =
            q + (bh * 2048 + mtile * 32 + g * 16 + lp) * 64;
#pragma unroll
        for (int f = 0; f < 2; ++f) {
            ((uint4*)&aq[g][f])[0] = *(const uint4*)(qrow + f * 32 + hf * 8);
            ((uint4*)&aq[g][f])[1] =
                *(const uint4*)(qrow + f * 32 + 16 + hf * 8);
        }
    }

    floatx8 o[2][4];
    float mrun[2][8], lrun[2][8];
#pragma unroll
    for (int g = 0; g < 2; ++g) {
#pragma unroll
        for (int t = 0; t < 4; ++t) o[g][t] = zero8();
#pragma unroll
        for (int j = 0; j < 8; ++j) {
            mrun[g][j] = -__builtin_inff();
            lrun[g][j] = 0.f;
        }
    }

    for (int kb = 0; kb < 2048; kb += 32) {
        floatx8 s[2][2];
#pragma unroll
        for (int g = 0; g < 2; ++g) { s[g][0] = zero8(); s[g][1] = zero8(); }

#pragma unroll
        for (int f = 0; f < 2; ++f) {
            // B = K^T: per-lane key n=lp, d-runs at half*8 and 16+half*8
            const unsigned short* kr0 =
                kmat + (bh * 2048 + kb + lp) * 64 + f * 32 + hf * 8;
            const unsigned short* kr1 = kr0 + 16 * 64;
            bf16x16 bk0, bk1;
            ((uint4*)&bk0)[0] = *(const uint4*)(kr0);
            ((uint4*)&bk0)[1] = *(const uint4*)(kr0 + 16);
            ((uint4*)&bk1)[0] = *(const uint4*)(kr1);
            ((uint4*)&bk1)[1] = *(const uint4*)(kr1 + 16);
#pragma unroll
            for (int g = 0; g < 2; ++g) {
                s[g][0] = __builtin_amdgcn_wmma_f32_16x16x32_bf16(
                    false, aq[g][f], false, bk0, (short)0, s[g][0], false,
                    false);
                s[g][1] = __builtin_amdgcn_wmma_f32_16x16x32_bf16(
                    false, aq[g][f], false, bk1, (short)0, s[g][1], false,
                    false);
            }
        }

        // online softmax over 32 keys; row m=j+8*hf lives in one 16-lane half
#pragma unroll
        for (int g = 0; g < 2; ++g)
#pragma unroll
            for (int j = 0; j < 8; ++j) {
                float r0 = s[g][0][j] * scale, r1 = s[g][1][j] * scale;
                float tm = fmaxf(r0, r1);
#pragma unroll
                for (int msk = 1; msk <= 8; msk <<= 1)
                    tm = fmaxf(tm, __shfl_xor(tm, msk, 32));
                float nm = fmaxf(mrun[g][j], tm);
                float al = __expf(mrun[g][j] - nm);
                float p0 = __expf(r0 - nm), p1 = __expf(r1 - nm);
                float rs = p0 + p1;
#pragma unroll
                for (int msk = 1; msk <= 8; msk <<= 1)
                    rs += __shfl_xor(rs, msk, 32);
                lrun[g][j] = lrun[g][j] * al + rs;
                mrun[g][j] = nm;
#pragma unroll
                for (int t = 0; t < 4; ++t) o[g][t][j] *= al;
                int m = j + 8 * hf;
                pbuf[g * 512 + m * 32 + lp]      = f2bf(p0);
                pbuf[g * 512 + m * 32 + 16 + lp] = f2bf(p1);
            }

        // intra-wave LDS RAW: DS is in-order, but fence + wait for safety
        asm volatile("s_wait_dscnt 0" ::: "memory");

        // P as A-fragments (16x32 each)
        bf16x16 pa[2];
#pragma unroll
        for (int g = 0; g < 2; ++g) {
            ((uint4*)&pa[g])[0] =
                *(const uint4*)&pbuf[g * 512 + lp * 32 + hf * 8];
            ((uint4*)&pa[g])[1] =
                *(const uint4*)&pbuf[g * 512 + lp * 32 + 16 + hf * 8];
        }

        // O += P @ V   (V pre-transposed: contiguous key-runs per d-row;
        //               each V fragment feeds both row groups)
#pragma unroll
        for (int t = 0; t < 4; ++t) {
            const unsigned short* vr =
                vT + (bh * 64 + t * 16 + lp) * 2048 + kb + hf * 8;
            bf16x16 bv;
            ((uint4*)&bv)[0] = *(const uint4*)(vr);
            ((uint4*)&bv)[1] = *(const uint4*)(vr + 16);
#pragma unroll
            for (int g = 0; g < 2; ++g)
                o[g][t] = __builtin_amdgcn_wmma_f32_16x16x32_bf16(
                    false, pa[g], false, bv, (short)0, o[g][t], false, false);
        }
        asm volatile("s_wait_dscnt 0" ::: "memory");
    }

    // epilogue: divide by row sum, store bf16 [B,N,C]
#pragma unroll
    for (int g = 0; g < 2; ++g)
#pragma unroll
        for (int j = 0; j < 8; ++j) {
            float inv = 1.0f / lrun[g][j];
            int row = mtile * 32 + g * 16 + j + 8 * hf;
#pragma unroll
            for (int t = 0; t < 4; ++t)
                y[((size_t)b * 2048 + row) * 768 + h * 64 + t * 16 + lp] =
                    f2bf(o[g][t][j] * inv);
        }
}

// ---------------------------------------------------------------- out proj
// y [8192][768] bf16 @ woutT[768][768] bf16 + bias -> fp32 out, 32x64 tiles
__global__ __launch_bounds__(32)
void out_gemm(const unsigned short* __restrict__ y,
              const unsigned short* __restrict__ wT,
              const float* __restrict__ bias, float* __restrict__ out) {
    const int lane = threadIdx.x;
    const int lp = lane & 15, hf = lane >> 4;
    const int mtile = blockIdx.x;   // 256
    const int ntile = blockIdx.y;   // 12
    const int K = 768;

    floatx8 acc[2][4];
#pragma unroll
    for (int g = 0; g < 2; ++g)
#pragma unroll
        for (int t = 0; t < 4; ++t) acc[g][t] = zero8();

    const unsigned short* arow0 = y + (size_t)(mtile * 32 + lp) * K;
    const unsigned short* arow1 = arow0 + (size_t)16 * K;
    const unsigned short* brow[4];
#pragma unroll
    for (int t = 0; t < 4; ++t)
        brow[t] = wT + (size_t)(ntile * 64 + t * 16 + lp) * K;

    for (int kk = 0; kk < K; kk += 32) {
        if (kk + 32 < K) {
            __builtin_prefetch(arow0 + kk + 32, 0, 0);
            __builtin_prefetch(arow1 + kk + 32, 0, 0);
        }
        bf16x16 a0, a1;
        ((uint4*)&a0)[0] = *(const uint4*)(arow0 + kk + hf * 8);
        ((uint4*)&a0)[1] = *(const uint4*)(arow0 + kk + 16 + hf * 8);
        ((uint4*)&a1)[0] = *(const uint4*)(arow1 + kk + hf * 8);
        ((uint4*)&a1)[1] = *(const uint4*)(arow1 + kk + 16 + hf * 8);
#pragma unroll
        for (int t = 0; t < 4; ++t) {
            bf16x16 bm;
            ((uint4*)&bm)[0] = *(const uint4*)(brow[t] + kk + hf * 8);
            ((uint4*)&bm)[1] = *(const uint4*)(brow[t] + kk + 16 + hf * 8);
            acc[0][t] = __builtin_amdgcn_wmma_f32_16x16x32_bf16(
                false, a0, false, bm, (short)0, acc[0][t], false, false);
            acc[1][t] = __builtin_amdgcn_wmma_f32_16x16x32_bf16(
                false, a1, false, bm, (short)0, acc[1][t], false, false);
        }
    }
#pragma unroll
    for (int g = 0; g < 2; ++g)
#pragma unroll
        for (int t = 0; t < 4; ++t)
#pragma unroll
            for (int j = 0; j < 8; ++j) {
                int m = mtile * 32 + g * 16 + j + 8 * hf;
                int c = ntile * 64 + t * 16 + lp;
                out[(size_t)m * 768 + c] = acc[g][t][j] + bias[c];
            }
}

// ---------------------------------------------------------------- launcher
extern "C" void kernel_launch(void* const* d_in, const int* in_sizes, int n_in,
                              void* d_out, int out_size, void* d_ws,
                              size_t ws_size, hipStream_t stream) {
    const float* x     = (const float*)d_in[0];
    const float* w_qkv = (const float*)d_in[1];
    const float* w_out = (const float*)d_in[2];
    const float* b_out = (const float*)d_in[3];
    float* out = (float*)d_out;

    const int B = 4, N = 2048, C = 768, H = 12;
    const size_t M = (size_t)B * N;        // 8192
    const size_t MC = M * C;               // 6.29M elems

    char* ws = (char*)d_ws;
    unsigned short* xb    = (unsigned short*)ws; ws += MC * 2;
    unsigned short* wqkvT = (unsigned short*)ws; ws += (size_t)3 * C * C * 2;
    unsigned short* woutT = (unsigned short*)ws; ws += (size_t)C * C * 2;
    unsigned short* qb    = (unsigned short*)ws; ws += MC * 2;
    unsigned short* kb    = (unsigned short*)ws; ws += MC * 2;
    unsigned short* vTb   = (unsigned short*)ws; ws += MC * 2;
    unsigned short* yb    = (unsigned short*)ws; ws += MC * 2;

    cvt_f32_bf16<<<2048, 256, 0, stream>>>(x, xb, (int)MC);
    transpose_cvt<<<2048, 256, 0, stream>>>(w_qkv, wqkvT, C, 3 * C);
    transpose_cvt<<<1024, 256, 0, stream>>>(w_out, woutT, C, C);

    qkv_gemm<<<dim3((unsigned)(M / 32), (3 * C) / 64), 32, 0, stream>>>(
        xb, wqkvT, qb, kb, vTb);
    flash_attn<<<dim3(N / 32, H, B), 32, 0, stream>>>(qb, kb, vTb, yb);
    out_gemm<<<dim3((unsigned)(M / 32), C / 64), 32, 0, stream>>>(
        yb, woutT, b_out, out);
}
